// ConcordanceIndex_14577119003163
// MI455X (gfx1250) — compile-verified, matrix operand order checked by reading, not run on hardware
//
#include <hip/hip_runtime.h>

typedef int v8i __attribute__((ext_vector_type(8)));

__global__ void ci_init_ws(unsigned int* ws) {
    if (threadIdx.x < 2) ws[threadIdx.x] = 0u;
}

// Pairwise concordance counting via V_WMMA_I32_16X16X64_IU8.
// Each wave owns a 16-row tile (rows = "a") and scans a 1024-wide slice of "b".
// A-matrix (16x64 u8): per-pair indicator (time_b > time_a) && (risk_a > risk_b).
// B-matrix: all-ones -> D accumulates per-row pair counts (all 16 columns equal).
__global__ __launch_bounds__(256) void ci_pairs(const float* __restrict__ risk,
                                                const float* __restrict__ timev,
                                                const float* __restrict__ eventv,
                                                unsigned int* __restrict__ ws,
                                                int sliceLen) {
    const int lane   = threadIdx.x & 31;
    const int wave   = threadIdx.x >> 5;
    const int hi     = lane >> 4;                       // which K half this lane supplies
    const int rowBase = ((blockIdx.x << 3) + wave) << 4; // 16 rows per wave
    const int row    = rowBase + (lane & 15);           // A-matrix row for this lane
    const int sliceBase = blockIdx.y * sliceLen;

    const float ta = timev[row];
    const float ra = risk[row];

    v8i acc = {};                 // i32 16x16 accumulator (8 VGPRs)
    v8i bones;
    #pragma unroll
    for (int g = 0; g < 8; ++g) bones[g] = 0x01010101;  // all-ones u8 B matrix

    unsigned int cntT = 0;        // #{b in my K-subset : time_b > time_a}

    const float4* t4 = reinterpret_cast<const float4*>(timev);
    const float4* r4 = reinterpret_cast<const float4*>(risk);

    const int ntiles = sliceLen >> 6;                   // K=64 per WMMA
    for (int kt = 0; kt < ntiles; ++kt) {
        const int b0 = sliceBase + (kt << 6);
        v8i a;
        #pragma unroll
        for (int v = 0; v < 8; ++v) {
            // ISA 8-bit A layout: VGPR v covers K = q*16 + r*4 + hi*8 .. +3
            const int q  = v >> 1;
            const int r  = v & 1;
            const int ks = b0 + q * 16 + r * 4 + hi * 8; // multiple of 4 -> b128 loads
            const float4 tb = t4[ks >> 2];
            const float4 rb = r4[ks >> 2];
            unsigned int packed = 0u;
            { const bool tg = tb.x > ta; cntT += tg; packed |= (unsigned int)(tg && (ra > rb.x));       }
            { const bool tg = tb.y > ta; cntT += tg; packed |= (unsigned int)(tg && (ra > rb.y)) << 8;  }
            { const bool tg = tb.z > ta; cntT += tg; packed |= (unsigned int)(tg && (ra > rb.z)) << 16; }
            { const bool tg = tb.w > ta; cntT += tg; packed |= (unsigned int)(tg && (ra > rb.w)) << 24; }
            a[v] = (int)packed;
        }
        // 7-arg iu8 form: (sgn_a, A, sgn_b, B, C, reuse_a, reuse_b)
        acc = __builtin_amdgcn_wmma_i32_16x16x64_iu8(false, a, false, bones, acc, false, false);
    }

    // den: pairs (a,b) with time_a < time_b, gated by event_a.
    // Lanes m and m+16 cover disjoint K halves -> each adds its own count once.
    if (eventv[row] > 0.0f) atomicAdd(&ws[1], cntT);

    // num: D-matrix VGPR g holds row (g + 8*hi); all 16 columns identical,
    // so only column-0 lanes (lane 0 and lane 16) reduce, weighted by event_a.
    if ((lane & 15) == 0) {
        unsigned int numPart = 0;
        #pragma unroll
        for (int g = 0; g < 8; ++g) {
            const int rr = rowBase + g + (hi << 3);
            if (eventv[rr] > 0.0f) numPart += (unsigned int)acc[g];
        }
        atomicAdd(&ws[0], numPart);
    }
}

__global__ void ci_final(const unsigned int* __restrict__ ws, float* __restrict__ out) {
    if (threadIdx.x == 0) {
        const unsigned int num = ws[0];
        const unsigned int den = ws[1];
        out[0] = (den != 0u) ? ((float)num / (float)den) : __builtin_nanf("");
    }
}

extern "C" void kernel_launch(void* const* d_in, const int* in_sizes, int n_in,
                              void* d_out, int out_size, void* d_ws, size_t ws_size,
                              hipStream_t stream) {
    (void)n_in; (void)out_size; (void)ws_size;
    const float* risk   = (const float*)d_in[0];
    const float* timev  = (const float*)d_in[1];
    const float* eventv = (const float*)d_in[2];
    float* out          = (float*)d_out;
    unsigned int* ws    = (unsigned int*)d_ws;

    const int n        = in_sizes[0];   // 8192
    const int slices   = 8;             // grid.y splits the b-range
    const int sliceLen = n / slices;    // 1024 -> 16 WMMA iterations per wave

    ci_init_ws<<<1, 32, 0, stream>>>(ws);

    // 8 waves per block, one 16-row tile per wave.
    dim3 grid(n / (16 * 8), slices);    // (64, 8)
    ci_pairs<<<grid, 256, 0, stream>>>(risk, timev, eventv, ws, sliceLen);

    ci_final<<<1, 32, 0, stream>>>(ws, out);
}
